// SRP_torch_46050639347978
// MI455X (gfx1250) — compile-verified, matrix operand order checked by reading, not run on hardware
//
#include <hip/hip_runtime.h>

// ---------------------------------------------------------------------------
// Sparse Random Projection: out[512,2048] = X[512,32768] * W^T,
// W = COO(rows, cols, vals) densified to f16, GEMM via v_wmma_f32_16x16x32_f16
// with double-buffered LDS tiles staged by GLOBAL_LOAD_ASYNC_TO_LDS_B128.
// ---------------------------------------------------------------------------

typedef __attribute__((ext_vector_type(16))) _Float16 v16h;
typedef __attribute__((ext_vector_type(8)))  _Float16 v8h;
typedef __attribute__((ext_vector_type(4)))  _Float16 v4h;
typedef __attribute__((ext_vector_type(8)))  float    v8f;
typedef __attribute__((ext_vector_type(4)))  float    f32x4;
typedef __attribute__((ext_vector_type(4)))  unsigned int u32x4;
typedef int gv4i __attribute__((vector_size(16)));   // matches builtin's pointee

#define N_COMP 2048
#define N_FEAT 32768
#define BATCH  512
#define SPLITK 8
#define KPER   (N_FEAT / SPLITK)   // 4096
#define TK     64
#define KITERS (KPER / TK)         // 64
#define LDSW   72                  // 64 + 8 halves pad (144B rows)

#define GLOBAL_AS __attribute__((address_space(1)))
#define LDS_AS    __attribute__((address_space(3)))

#if defined(__AMDGCN__) && \
    __has_builtin(__builtin_amdgcn_global_load_async_to_lds_b128) && \
    __has_builtin(__builtin_amdgcn_s_wait_asynccnt)
#define HAS_ASYNC 1
#else
#define HAS_ASYNC 0
#endif

// Stage 16 bytes global -> LDS. Async path uses the CDNA5 ASYNCcnt-tracked
// direct-to-LDS copy; fallback goes through VGPRs + ds_store_b128.
__device__ __forceinline__ void stage16(const _Float16* g, _Float16* l) {
#if HAS_ASYNC
    __builtin_amdgcn_global_load_async_to_lds_b128(
        (GLOBAL_AS gv4i*)g, (LDS_AS gv4i*)l, 0, 0);
#else
    *(u32x4*)l = *(const u32x4*)g;
#endif
}

__device__ __forceinline__ void wait_async() {
#if HAS_ASYNC
    __builtin_amdgcn_s_wait_asynccnt(0);
#endif
}

// ---------------- helper kernels -------------------------------------------

__global__ void zero_b128(u32x4* __restrict__ p, long n4) {
    long i = (long)blockIdx.x * blockDim.x + threadIdx.x;
    if (i < n4) {
        u32x4 z = {0u, 0u, 0u, 0u};
        p[i] = z;
    }
}

__global__ void cvt_f32_f16(const f32x4* __restrict__ X, v4h* __restrict__ Xh, int n4) {
    int i = blockIdx.x * blockDim.x + threadIdx.x;
    if (i < n4) {
        f32x4 x = X[i];
        Xh[i] = __builtin_convertvector(x, v4h);
    }
}

// COO scatter-add into dense f16 W (duplicates must accumulate -> CAS on the
// containing 32-bit word).
__global__ void scatter_vals(const float* __restrict__ vals,
                             const int*   __restrict__ rows,
                             const int*   __restrict__ cols,
                             unsigned int* __restrict__ W, int nnz) {
    int k = blockIdx.x * blockDim.x + threadIdx.x;
    if (k >= nnz) return;
    unsigned idx  = (unsigned)rows[k] * (unsigned)N_FEAT + (unsigned)cols[k];
    unsigned word = idx >> 1;
    unsigned sh   = (idx & 1u) * 16u;
    _Float16 v    = (_Float16)vals[k];
    unsigned int* p = W + word;
    unsigned old = *p, assumed;
    do {
        assumed = old;
        unsigned short curb = (unsigned short)((assumed >> sh) & 0xFFFFu);
        _Float16 cur = __builtin_bit_cast(_Float16, curb);
        _Float16 nv  = cur + v;
        unsigned short nb = __builtin_bit_cast(unsigned short, nv);
        unsigned neww = (assumed & ~(0xFFFFu << sh)) | ((unsigned)nb << sh);
        old = atomicCAS(p, assumed, neww);
    } while (old != assumed);
}

// ---------------- WMMA GEMM ------------------------------------------------
// grid: (N_COMP/128, BATCH/128, SPLITK), block: 256 threads (8 wave32s, 2x4).
// Each wave computes a 64(M) x 32(N) output tile = 4x2 wmma 16x16 tiles.
// Double-buffered LDS: async-copy tile k+1 while doing WMMAs on tile k.

__global__ __launch_bounds__(256)
void srp_gemm(const _Float16* __restrict__ Xh,   // [BATCH][N_FEAT] f16
              const _Float16* __restrict__ W,    // [N_COMP][N_FEAT] f16
              float* __restrict__ out)           // [BATCH][N_COMP] f32 (atomic)
{
    __shared__ __align__(16) _Float16 lA[2][128 * LDSW];
    __shared__ __align__(16) _Float16 lB[2][128 * LDSW];

    const int tid  = threadIdx.x;
    const int lane = tid & 31;
    const int wv   = tid >> 5;
    const int wm   = wv >> 2;        // 0..1
    const int wn   = wv & 3;         // 0..3
    const int lh   = lane >> 4;      // lane-half selector
    const int lm   = lane & 15;

    const int m0 = blockIdx.y * 128;              // batch-row base
    const int n0 = blockIdx.x * 128;              // component base
    const size_t k0 = (size_t)blockIdx.z * KPER;  // split-K base

    // Per-thread staging coordinates: 1024 16B chunks per matrix per K-tile;
    // this thread owns chunks {tid + i*256} -> rows {row + i*32}, 16B slot cv.
    const int row = tid >> 3;        // 0..31
    const int cv  = tid & 7;
    const _Float16* gA = Xh + (size_t)(m0 + row) * N_FEAT + cv * 8 + k0;
    const _Float16* gB = W  + (size_t)(n0 + row) * N_FEAT + cv * 8 + k0;
    const int sOff = row * LDSW + cv * 8;

    auto stage = [&](int bi, int kb) {
#pragma unroll
        for (int i = 0; i < 4; ++i) {
            stage16(gA + kb + (size_t)i * 32 * N_FEAT, &lA[bi][sOff + i * 32 * LDSW]);
            stage16(gB + kb + (size_t)i * 32 * N_FEAT, &lB[bi][sOff + i * 32 * LDSW]);
        }
    };

    const v8f vzero = {0.f, 0.f, 0.f, 0.f, 0.f, 0.f, 0.f, 0.f};
    v8f acc[4][2];
#pragma unroll
    for (int m = 0; m < 4; ++m)
#pragma unroll
        for (int n = 0; n < 2; ++n) acc[m][n] = vzero;

    stage(0, 0);   // prologue: fill buffer 0

#pragma unroll 2
    for (int kt = 0; kt < KITERS; ++kt) {
        const int cur = kt & 1;
        wait_async();          // my async writes into lX[cur] are done
        __syncthreads();       // all writes visible + prev reads of lX[cur^1] done
        if (kt + 1 < KITERS) stage(cur ^ 1, (kt + 1) * TK);

        const _Float16* A = lA[cur];
        const _Float16* B = lB[cur];
#pragma unroll
        for (int kk = 0; kk < 2; ++kk) {
            // Fragment layout per ISA 7.12.2 (16-bit A 16x32):
            //   lanes 0-15 : M=lane, K = {0..7, 16..23}
            //   lanes 16-31: M=lane-16, K = {8..15, 24..31}
            union Frag { v16h v; v8h h[2]; };
            Frag a[4], b[2];
#pragma unroll
            for (int m = 0; m < 4; ++m) {
                const _Float16* p = A + (wm * 64 + m * 16 + lm) * LDSW + kk * 32 + lh * 8;
                a[m].h[0] = *(const v8h*)p;
                a[m].h[1] = *(const v8h*)(p + 16);
            }
#pragma unroll
            for (int n = 0; n < 2; ++n) {
                const _Float16* p = B + (wn * 32 + n * 16 + lm) * LDSW + kk * 32 + lh * 8;
                b[n].h[0] = *(const v8h*)p;
                b[n].h[1] = *(const v8h*)(p + 16);
            }
#pragma unroll
            for (int m = 0; m < 4; ++m)
#pragma unroll
                for (int n = 0; n < 2; ++n)
                    acc[m][n] = __builtin_amdgcn_wmma_f32_16x16x32_f16(
                        false, a[m].v, false, b[n].v,
                        (short)0, acc[m][n], false, false);
        }
    }

    // Epilogue: C/D layout -> VGPR j holds M=j (lanes 0-15) / M=j+8 (16-31),
    // N = lane&15. Accumulate split-K partials with f32 atomics.
    const int Mb = m0 + wm * 64;
    const int Nb = n0 + wn * 32;
#pragma unroll
    for (int m = 0; m < 4; ++m)
#pragma unroll
        for (int n = 0; n < 2; ++n)
#pragma unroll
            for (int j = 0; j < 8; ++j) {
                int M = Mb + m * 16 + j + 8 * lh;
                int N = Nb + n * 16 + lm;
                atomicAdd(out + (size_t)M * N_COMP + N, acc[m][n][j]);
            }
}

// Fallback if workspace is too small: direct COO gather + atomic scatter.
__global__ void srp_fallback(const float* __restrict__ X,
                             const float* __restrict__ vals,
                             const int*   __restrict__ rows,
                             const int*   __restrict__ cols,
                             float* __restrict__ out, int nnz) {
    int k = blockIdx.x;
    if (k >= nnz) return;
    int b = blockIdx.y * blockDim.x + threadIdx.x;
    float v = vals[k];
    int r = rows[k], c = cols[k];
    atomicAdd(out + (size_t)b * N_COMP + r, v * X[(size_t)b * N_FEAT + c]);
}

// ---------------- host launcher --------------------------------------------

extern "C" void kernel_launch(void* const* d_in, const int* in_sizes, int n_in,
                              void* d_out, int out_size, void* d_ws, size_t ws_size,
                              hipStream_t stream) {
    const float* X    = (const float*)d_in[0];
    const float* vals = (const float*)d_in[1];
    const int*   rows = (const int*)d_in[2];
    const int*   cols = (const int*)d_in[3];
    float* out = (float*)d_out;
    const int nnz = in_sizes[1];

    // Output is accumulated atomically -> must be zeroed every call
    {
        long n4 = (long)out_size / 4;
        zero_b128<<<(int)((n4 + 255) / 256), 256, 0, stream>>>((u32x4*)d_out, n4);
    }

    const size_t Wbytes  = (size_t)N_COMP * N_FEAT * sizeof(_Float16); // 128 MB
    const size_t XhBytes = (size_t)BATCH  * N_FEAT * sizeof(_Float16); //  32 MB

    if (ws_size >= Wbytes + XhBytes) {
        _Float16* W  = (_Float16*)d_ws;
        _Float16* Xh = (_Float16*)((char*)d_ws + Wbytes);

        long wn4 = (long)(Wbytes / 16);
        zero_b128<<<(int)((wn4 + 255) / 256), 256, 0, stream>>>((u32x4*)W, wn4);

        int n4 = BATCH * N_FEAT / 4;
        cvt_f32_f16<<<(n4 + 255) / 256, 256, 0, stream>>>((const f32x4*)X, (v4h*)Xh, n4);

        scatter_vals<<<(nnz + 255) / 256, 256, 0, stream>>>(vals, rows, cols,
                                                            (unsigned int*)W, nnz);

        dim3 grid(N_COMP / 128, BATCH / 128, SPLITK);
        srp_gemm<<<grid, 256, 0, stream>>>(Xh, W, out);
    } else {
        dim3 grid(nnz, BATCH / 256);
        srp_fallback<<<grid, 256, 0, stream>>>(X, vals, rows, cols, out, nnz);
    }
}
